// GatedDeltaNetDecoderLayer_59528246722589
// MI455X (gfx1250) — compile-verified, matrix operand order checked by reading, not run on hardware
//
#include <hip/hip_runtime.h>
#include <hip/hip_bf16.h>

// ---------------- shapes (fixed by reference) ----------------
#define T_SEQ     4096
#define D_MODEL   2048
#define H_HEADS   16
#define DK        128
#define M_FFN     8192
#define QKVZ_COLS 8192   // 2*kd + 2*vd
#define QKV_COLS  6144   // q|k|v conv channels
#define CHUNKC    64
#define NCHUNK    (T_SEQ / CHUNKC)
#define DVB       32     // Dv slice per scan block

// Detect the TDM builtin; fall back to cooperative staging.
#if defined(__has_builtin)
#if __has_builtin(__builtin_amdgcn_tensor_load_to_lds) && __has_builtin(__builtin_amdgcn_s_wait_tensorcnt)
#define USE_TDM 1
#else
#define USE_TDM 0
#endif
#else
#define USE_TDM 0
#endif

typedef __attribute__((ext_vector_type(16))) __bf16 v16bf;
typedef __attribute__((ext_vector_type(8)))  float  v8f;

__device__ __forceinline__ __bf16 to_bf16(float f) {
  union { __hip_bfloat16 h; __bf16 b; } u;
  u.h = __float2bfloat16(f);
  return u.b;
}

__device__ __forceinline__ v8f wmma_bf16(v16bf a, v16bf b, v8f c) {
  // emits v_wmma_f32_16x16x32_bf16
  return __builtin_amdgcn_wmma_f32_16x16x32_bf16(false, a, false, b, (short)0, c, false, false);
}

// A/B operand fragment from contiguous bf16 row-major tile (16 rows x 32 K), ld in elements.
// Per CDNA5 ISA: lane<16 holds row=lane, K {0..7,16..23}; lane>=16 row=lane-16, K {8..15,24..31}.
__device__ __forceinline__ v16bf load_frag_bf16(const __hip_bfloat16* p, int ld) {
  int lane = threadIdx.x & 31;
  int r = lane & 15, half = lane >> 4;
  const __hip_bfloat16* row = p + (size_t)r * ld;
  union { uint4 q[2]; v16bf v; } u;
  u.q[0] = *reinterpret_cast<const uint4*>(row + half * 8);
  u.q[1] = *reinterpret_cast<const uint4*>(row + 16 + half * 8);
  return u.v;
}

// Generic gather fragment builder: f(r, k) -> float for a 16x32 tile.
template <typename F>
__device__ __forceinline__ v16bf make_frag(F f) {
  int lane = threadIdx.x & 31;
  int r = lane & 15, half = lane >> 4;
  v16bf v;
#pragma unroll
  for (int e = 0; e < 16; ++e) {
    int kk = ((e & 8) ? 16 : 0) + half * 8 + (e & 7);
    v[e] = to_bf16(f(r, kk));
  }
  return v;
}

__device__ __forceinline__ float silu_f(float x) { return x / (1.f + __expf(-x)); }
__device__ __forceinline__ float softplus_f(float x) {
  return (x > 20.f) ? x : log1pf(__expf(x));
}

// ---------------- elementwise utilities ----------------
__global__ void f32_to_bf16_kernel(const float* __restrict__ in,
                                   __hip_bfloat16* __restrict__ out, size_t n) {
  size_t i = (size_t)blockIdx.x * blockDim.x + threadIdx.x;
  size_t stride = (size_t)gridDim.x * blockDim.x;
  for (; i < n; i += stride) out[i] = __float2bfloat16(in[i]);
}

__global__ void add_f32_kernel(const float* __restrict__ a, const float* __restrict__ b,
                               float* __restrict__ out, size_t n) {
  size_t i = (size_t)blockIdx.x * blockDim.x + threadIdx.x;
  size_t stride = (size_t)gridDim.x * blockDim.x;
  for (; i < n; i += stride) out[i] = a[i] + b[i];
}

__global__ void silu_mul_bf16_kernel(const float* __restrict__ g, const float* __restrict__ u,
                                     __hip_bfloat16* __restrict__ out, size_t n) {
  size_t i = (size_t)blockIdx.x * blockDim.x + threadIdx.x;
  size_t stride = (size_t)gridDim.x * blockDim.x;
  for (; i < n; i += stride) out[i] = __float2bfloat16(silu_f(g[i]) * u[i]);
}

// ---------------- RMSNorm -> bf16 ----------------
__global__ void __launch_bounds__(256)
rmsnorm_bf16_kernel(const float* __restrict__ x, const float* __restrict__ w,
                    __hip_bfloat16* __restrict__ out, int D) {
  __shared__ float red[256];
  int t = blockIdx.x;
  const float* xr = x + (size_t)t * D;
  float s = 0.f;
  for (int i = threadIdx.x; i < D; i += 256) { float v = xr[i]; s += v * v; }
  red[threadIdx.x] = s;
  __syncthreads();
  for (int st = 128; st > 0; st >>= 1) {
    if (threadIdx.x < st) red[threadIdx.x] += red[threadIdx.x + st];
    __syncthreads();
  }
  float inv = rsqrtf(red[0] / (float)D + 1e-5f);
  for (int i = threadIdx.x; i < D; i += 256)
    out[(size_t)t * D + i] = __float2bfloat16(xr[i] * inv * w[i]);
}

// ---------------- WMMA NT GEMM with TDM/LDS double buffering ----------------
// C[M,N] = A[M,K] * B[N,K]^T. Block tile 128x128, K-step 32.
// 8 waves as 4(m) x 2(n); wave tile 32x64 (2x4 accumulators).
#define BM 128
#define BN 128
#define BK 32
#define LDS_ROW 40  // 32 elements + 8-element (16B) TDM pad -> conflict-free ds_load_b128

#if USE_TDM
// Issue one TDM 2D tile load: `rows` x 32 bf16, row stride ld_elems, packed into LDS
// with 16B padding after each 64B row (pad_interval=16 DW, pad_amount=4 DW).
__device__ __forceinline__ void tdm_load_tile(const __hip_bfloat16* gsrc, int ld_elems,
                                              __hip_bfloat16* lds_dst, int rows) {
  typedef unsigned int v4u __attribute__((ext_vector_type(4)));
  typedef int v8i_t __attribute__((ext_vector_type(8)));
  typedef int v4i_t __attribute__((ext_vector_type(4)));
  unsigned long long ga = (unsigned long long)(size_t)gsrc;
  unsigned int laddr = (unsigned int)(size_t)lds_dst;  // low 32 bits = LDS byte offset
  v4u g0;
  g0[0] = 1u;                                   // count=1, user descriptor
  g0[1] = laddr;                                // lds_addr
  g0[2] = (unsigned int)ga;                     // global_addr[31:0]
  g0[3] = (unsigned int)((ga >> 32) & 0x1FFFFFFull) | (2u << 30);  // addr[56:32] | type=2
  v8i_t g1;
  g1[0] = (int)((1u << 16)      // data_size = 2 bytes
              | (1u << 20)      // pad_enable
              | (3u << 22)      // pad_interval: 16 dwords (64B)
              | (3u << 25));    // pad_amount: 4 dwords (16B)
  g1[1] = (int)(32u << 16);                     // tensor_dim0 = 32 (lo16 in hi half)
  g1[2] = (int)((unsigned)rows << 16);          // tensor_dim0 hi=0 | tensor_dim1 lo = rows
  g1[3] = (int)(32u << 16);                     // tensor_dim1 hi=0 | tile_dim0 = 32
  g1[4] = (int)((unsigned)rows & 0xFFFFu);      // tile_dim1 = rows | tile_dim2 = 0
  g1[5] = (int)ld_elems;                        // tensor_dim0_stride lo
  g1[6] = 0;                                    // stride hi | tensor_dim1_stride lo
  g1[7] = 0;
  v4i_t z4 = {0, 0, 0, 0};
  v8i_t z8 = {0, 0, 0, 0, 0, 0, 0, 0};
  // clang-23 / therock toolchain: 6-arg form (g0, g1, g2, g3, g4, cpol)
  __builtin_amdgcn_tensor_load_to_lds(g0, g1, z4, z4, z8, 0);
}
#else
// Cooperative fallback: 256 threads stage a 128x32 bf16 tile into padded LDS.
__device__ __forceinline__ void stage_tile_128x32(const __hip_bfloat16* g, int ldg,
                                                  __hip_bfloat16* lds) {
  int t = threadIdx.x;
  int r = t >> 1, h16 = (t & 1) * 16;
  const uint4* src = reinterpret_cast<const uint4*>(g + (size_t)r * ldg + h16);
  uint4 v0 = src[0];
  uint4 v1 = src[1];
  uint4* dst = reinterpret_cast<uint4*>(lds + r * LDS_ROW + h16);
  dst[0] = v0;
  dst[1] = v1;
}
#endif

__device__ __forceinline__ void gemm_compute_step(const __hip_bfloat16* As,
                                                  const __hip_bfloat16* Bs,
                                                  int wm, int wn, v8f (&acc)[2][4]) {
  v16bf a[2], b[4];
#pragma unroll
  for (int i = 0; i < 2; ++i) a[i] = load_frag_bf16(As + (wm * 32 + i * 16) * LDS_ROW, LDS_ROW);
#pragma unroll
  for (int j = 0; j < 4; ++j) b[j] = load_frag_bf16(Bs + (wn * 64 + j * 16) * LDS_ROW, LDS_ROW);
#pragma unroll
  for (int i = 0; i < 2; ++i)
#pragma unroll
    for (int j = 0; j < 4; ++j) acc[i][j] = wmma_bf16(a[i], b[j], acc[i][j]);
}

__global__ void __launch_bounds__(256)
gemm_bf16_nt_kernel(const __hip_bfloat16* __restrict__ A, const __hip_bfloat16* __restrict__ B,
                    float* __restrict__ C, int M, int N, int K) {
  __shared__ __align__(16) __hip_bfloat16 Asmem[2][BM * LDS_ROW];
  __shared__ __align__(16) __hip_bfloat16 Bsmem[2][BN * LDS_ROW];
  int tid = threadIdx.x;
  int wid = tid >> 5, lane = tid & 31;
  int wm = wid >> 1, wn = wid & 1;
  int mb = blockIdx.y * BM, nb = blockIdx.x * BN;
  v8f zero = {0, 0, 0, 0, 0, 0, 0, 0};
  v8f acc[2][4];
#pragma unroll
  for (int i = 0; i < 2; ++i)
#pragma unroll
    for (int j = 0; j < 4; ++j) acc[i][j] = zero;
  int nsteps = K / BK;

#if USE_TDM
  if (wid == 0) {
    tdm_load_tile(A + (size_t)mb * K, K, Asmem[0], BM);
    tdm_load_tile(B + (size_t)nb * K, K, Bsmem[0], BN);
    __builtin_amdgcn_s_wait_tensorcnt((short)0);
  }
  __syncthreads();
  for (int s = 0; s < nsteps; ++s) {
    int cur = s & 1;
    if (wid == 0 && s + 1 < nsteps) {
      size_t ko = (size_t)(s + 1) * BK;
      tdm_load_tile(A + (size_t)mb * K + ko, K, Asmem[cur ^ 1], BM);
      tdm_load_tile(B + (size_t)nb * K + ko, K, Bsmem[cur ^ 1], BN);
    }
    gemm_compute_step(Asmem[cur], Bsmem[cur], wm, wn, acc);
    if (wid == 0) __builtin_amdgcn_s_wait_tensorcnt((short)0);
    __syncthreads();
  }
#else
  for (int s = 0; s < nsteps; ++s) {
    __syncthreads();
    stage_tile_128x32(A + (size_t)mb * K + (size_t)s * BK, K, Asmem[0]);
    stage_tile_128x32(B + (size_t)nb * K + (size_t)s * BK, K, Bsmem[0]);
    __syncthreads();
    gemm_compute_step(Asmem[0], Bsmem[0], wm, wn, acc);
  }
#endif

  int half = lane >> 4, col = lane & 15;
#pragma unroll
  for (int i = 0; i < 2; ++i)
#pragma unroll
    for (int j = 0; j < 4; ++j)
#pragma unroll
      for (int r = 0; r < 8; ++r) {
        int mm = mb + wm * 32 + i * 16 + r + 8 * half;
        int nn = nb + wn * 64 + j * 16 + col;
        C[(size_t)mm * N + nn] = acc[i][j][r];
      }
}

// ---------------- b/a projection + gating scalars ----------------
__global__ void __launch_bounds__(256)
ba_gate_kernel(const __hip_bfloat16* __restrict__ h, const float* __restrict__ ba_w,
               const float* __restrict__ A_log, const float* __restrict__ dt_bias,
               const float* __restrict__ mask, float* __restrict__ g_out,
               float* __restrict__ beta_out) {
  __shared__ float red[256];
  int t = blockIdx.x;
  int j = threadIdx.x & 31, sl = threadIdx.x >> 5;
  const __hip_bfloat16* hr = h + (size_t)t * D_MODEL;
  const float* wr = ba_w + (size_t)j * D_MODEL;
  float partial = 0.f;
  for (int k = sl; k < D_MODEL; k += 8) partial += __bfloat162float(hr[k]) * wr[k];
  red[threadIdx.x] = partial;
  __syncthreads();
  if (threadIdx.x < 32) {
    float s = 0.f;
#pragma unroll
    for (int p = 0; p < 8; ++p) s += red[p * 32 + threadIdx.x];
    float mk = mask[t];
    if (threadIdx.x < H_HEADS) {
      beta_out[(size_t)t * H_HEADS + threadIdx.x] = mk / (1.f + __expf(-s));
    } else {
      int hh = threadIdx.x - H_HEADS;
      g_out[(size_t)t * H_HEADS + hh] =
          -__expf(A_log[hh]) * softplus_f(s + dt_bias[hh]) * mk;
    }
  }
}

// ---------------- causal conv + SiLU + per-head l2norm (q scaled) ----------------
__global__ void __launch_bounds__(256)
conv_norm_kernel(const float* __restrict__ qkvz, const float* __restrict__ conv_w,
                 const float* __restrict__ mask, float* __restrict__ qkv) {
  __shared__ float row[QKV_COLS];
  __shared__ float gsum[32][8];
  __shared__ float gscale[32];
  int t = blockIdx.x;
  for (int c = threadIdx.x; c < QKV_COLS; c += 256) {
    float acc = 0.f;
#pragma unroll
    for (int j = 0; j < 4; ++j) {
      int tt = t - 3 + j;
      if (tt >= 0) acc += qkvz[(size_t)tt * QKVZ_COLS + c] * mask[tt] * conv_w[c * 4 + j];
    }
    row[c] = silu_f(acc);
  }
  __syncthreads();
  int gi = threadIdx.x >> 3, lt = threadIdx.x & 7;   // 32 groups (16 q-heads, 16 k-heads)
  int base = gi * DK;
  float ss = 0.f;
  for (int d = lt; d < DK; d += 8) { float v = row[base + d]; ss += v * v; }
  gsum[gi][lt] = ss;
  __syncthreads();
  if (lt == 0) {
    float tot = 0.f;
#pragma unroll
    for (int p = 0; p < 8; ++p) tot += gsum[gi][p];
    float sc = rsqrtf(tot + 1e-6f);
    if (gi < 16) sc *= 0.088388347648318447f;  // Dk^-0.5 folded into q
    gscale[gi] = sc;
  }
  __syncthreads();
  size_t obase = (size_t)t * QKV_COLS;
  for (int c = threadIdx.x; c < QKV_COLS; c += 256) {
    float v = row[c];
    if (c < 4096) v *= gscale[c >> 7];
    qkv[obase + c] = v;
  }
}

// ---------------- per-chunk prep: A, (I+A)^-1, u, w ----------------
__global__ void __launch_bounds__(256)
chunk_prep_kernel(const float* __restrict__ qkv, const float* __restrict__ g,
                  const float* __restrict__ beta, float* __restrict__ gcs_out,
                  float* __restrict__ u_out, float* __restrict__ w_out) {
  __shared__ float gcs_s[CHUNKC];
  __shared__ float beta_s[CHUNKC];
  __shared__ float A_s[CHUNKC][CHUNKC + 1];
  __shared__ float T_s[CHUNKC][CHUNKC + 1];
  int n = blockIdx.x, h = blockIdx.y;
  int t0 = n * CHUNKC;
  int tid = threadIdx.x;
  int wid = tid >> 5, lane = tid & 31, half = lane >> 4, col = lane & 15;

  if (tid < CHUNKC) beta_s[tid] = beta[(size_t)(t0 + tid) * H_HEADS + h];
  if (tid == 0) {
    float acc = 0.f;
    for (int c = 0; c < CHUNKC; ++c) {
      acc += g[(size_t)(t0 + c) * H_HEADS + h];
      gcs_s[c] = acc;
    }
  }
  for (int i = tid; i < CHUNKC * CHUNKC; i += 256) {
    int r = i >> 6, c = i & 63;
    T_s[r][c] = (r == c) ? 1.f : 0.f;
  }
  __syncthreads();
  if (tid < CHUNKC) gcs_out[(size_t)h * T_SEQ + t0 + tid] = gcs_s[tid];

  const float* kbase = qkv + 2048 + h * DK;  // (t,d) at kbase[t*QKV_COLS + d]
  const float* vbase = qkv + 4096 + h * DK;

  // A = (k_beta @ k^T) * decay, strictly lower. 16 tiles of 16x16.
  for (int tt2 = wid; tt2 < 16; tt2 += 8) {
    int m0 = (tt2 >> 2) * 16, n0 = (tt2 & 3) * 16;
    v8f acc = {0, 0, 0, 0, 0, 0, 0, 0};
    for (int k0 = 0; k0 < DK; k0 += 32) {
      v16bf af = make_frag([&](int r, int kk) {
        return kbase[(size_t)(t0 + m0 + r) * QKV_COLS + k0 + kk] * beta_s[m0 + r];
      });
      v16bf bf = make_frag([&](int nn, int kk) {
        return kbase[(size_t)(t0 + n0 + nn) * QKV_COLS + k0 + kk];
      });
      acc = wmma_bf16(af, bf, acc);
    }
#pragma unroll
    for (int j = 0; j < 8; ++j) {
      int rr = m0 + j + 8 * half, cc = n0 + col;
      A_s[rr][cc] = (rr > cc) ? acc[j] * __expf(gcs_s[rr] - gcs_s[cc]) : 0.f;
    }
  }
  __syncthreads();

  // forward substitution: T = (I + A)^-1 (unit lower triangular)
  for (int i = 1; i < CHUNKC; ++i) {
    if (tid < i) {
      float s = A_s[i][tid];
      for (int l = tid + 1; l < i; ++l) s += A_s[i][l] * T_s[l][tid];
      T_s[i][tid] = -s;
    }
    __syncthreads();
  }

  // u = T @ (v*beta), w = T @ (k*beta*exp(gcs)); 32 tiles, K=64
  for (int tt2 = wid; tt2 < 32; tt2 += 8) {
    int m0 = (tt2 >> 3) * 16, n0 = (tt2 & 7) * 16;
    v8f accu = {0, 0, 0, 0, 0, 0, 0, 0}, accw = {0, 0, 0, 0, 0, 0, 0, 0};
    for (int k0 = 0; k0 < CHUNKC; k0 += 32) {
      v16bf tf = make_frag([&](int r, int kk) { return T_s[m0 + r][k0 + kk]; });
      v16bf vb = make_frag([&](int nn, int kk) {
        int c = k0 + kk;
        return vbase[(size_t)(t0 + c) * QKV_COLS + n0 + nn] * beta_s[c];
      });
      accu = wmma_bf16(tf, vb, accu);
      v16bf kb = make_frag([&](int nn, int kk) {
        int c = k0 + kk;
        return kbase[(size_t)(t0 + c) * QKV_COLS + n0 + nn] * beta_s[c] * __expf(gcs_s[c]);
      });
      accw = wmma_bf16(tf, kb, accw);
    }
#pragma unroll
    for (int j = 0; j < 8; ++j) {
      int rr = m0 + j + 8 * half;
      size_t offo = ((size_t)h * T_SEQ + t0 + rr) * DK + n0 + col;
      u_out[offo] = accu[j];
      w_out[offo] = accw[j];
    }
  }
}

// ---------------- inter-chunk scan: one (head, Dv-slice) per block ----------------
__global__ void __launch_bounds__(256)
scan_kernel(const float* __restrict__ qkv, const float* __restrict__ gcs,
            const float* __restrict__ u_ws, const float* __restrict__ w_ws,
            float* __restrict__ o_out) {
  extern __shared__ float smem[];
  float* S = smem;                         // [128][DVB]
  float* vn = S + 128 * DVB;               // [64][DVB]
  float* attn = vn + CHUNKC * DVB;         // [64][64]
  float* gcs_s = attn + CHUNKC * CHUNKC;   // [64]
  int h = blockIdx.x;
  int dv0 = blockIdx.y * DVB;
  int tid = threadIdx.x, wid = tid >> 5, lane = tid & 31, half = lane >> 4, col = lane & 15;
  const float* qbase = qkv + 0 + h * DK;
  const float* kbase = qkv + 2048 + h * DK;
  const int NTV = DVB / 16;  // n-tiles over the Dv slice (=2)
  for (int i = tid; i < 128 * DVB; i += 256) S[i] = 0.f;
  __syncthreads();

  for (int n = 0; n < NCHUNK; ++n) {
    int t0 = n * CHUNKC;
    if (tid < CHUNKC) gcs_s[tid] = gcs[(size_t)h * T_SEQ + t0 + tid];
    __syncthreads();
    float g_last = gcs_s[CHUNKC - 1];

    // attn = (q k^T) * decay, lower (incl diag); 16 tiles
    for (int tt2 = wid; tt2 < 16; tt2 += 8) {
      int m0 = (tt2 >> 2) * 16, n0 = (tt2 & 3) * 16;
      v8f acc = {0, 0, 0, 0, 0, 0, 0, 0};
      for (int k0 = 0; k0 < DK; k0 += 32) {
        v16bf af = make_frag([&](int r, int kk) {
          return qbase[(size_t)(t0 + m0 + r) * QKV_COLS + k0 + kk];
        });
        v16bf bf = make_frag([&](int nn, int kk) {
          return kbase[(size_t)(t0 + n0 + nn) * QKV_COLS + k0 + kk];
        });
        acc = wmma_bf16(af, bf, acc);
      }
#pragma unroll
      for (int j = 0; j < 8; ++j) {
        int rr = m0 + j + 8 * half, cc = n0 + col;
        attn[rr * CHUNKC + cc] = (rr >= cc) ? acc[j] * __expf(gcs_s[rr] - gcs_s[cc]) : 0.f;
      }
    }
    __syncthreads();

    // v_new = u - w @ S ; 4 x NTV tiles
    for (int tt2 = wid; tt2 < 4 * NTV; tt2 += 8) {
      int m0 = (tt2 / NTV) * 16, n0 = (tt2 % NTV) * 16;
      v8f acc = {0, 0, 0, 0, 0, 0, 0, 0};
      for (int k0 = 0; k0 < DK; k0 += 32) {
        v16bf wf = make_frag([&](int r, int kk) {
          return w_ws[((size_t)h * T_SEQ + t0 + m0 + r) * DK + k0 + kk];
        });
        v16bf sf = make_frag([&](int nn, int kk) { return S[(k0 + kk) * DVB + n0 + nn]; });
        acc = wmma_bf16(wf, sf, acc);
      }
#pragma unroll
      for (int j = 0; j < 8; ++j) {
        int rr = m0 + j + 8 * half, cc = n0 + col;
        vn[rr * DVB + cc] = u_ws[((size_t)h * T_SEQ + t0 + rr) * DK + dv0 + cc] - acc[j];
      }
    }
    __syncthreads();

    // o = (q * exp(gcs)) @ S + attn @ v_new ; 4 x NTV tiles
    for (int tt2 = wid; tt2 < 4 * NTV; tt2 += 8) {
      int m0 = (tt2 / NTV) * 16, n0 = (tt2 % NTV) * 16;
      v8f acc = {0, 0, 0, 0, 0, 0, 0, 0};
      for (int k0 = 0; k0 < DK; k0 += 32) {
        v16bf qf = make_frag([&](int r, int kk) {
          return qbase[(size_t)(t0 + m0 + r) * QKV_COLS + k0 + kk] * __expf(gcs_s[m0 + r]);
        });
        v16bf sf = make_frag([&](int nn, int kk) { return S[(k0 + kk) * DVB + n0 + nn]; });
        acc = wmma_bf16(qf, sf, acc);
      }
      for (int k0 = 0; k0 < CHUNKC; k0 += 32) {
        v16bf af = make_frag([&](int r, int kk) { return attn[(m0 + r) * CHUNKC + k0 + kk]; });
        v16bf vf = make_frag([&](int nn, int kk) { return vn[(k0 + kk) * DVB + n0 + nn]; });
        acc = wmma_bf16(af, vf, acc);
      }
#pragma unroll
      for (int j = 0; j < 8; ++j) {
        int rr = m0 + j + 8 * half, cc = n0 + col;
        o_out[(size_t)(t0 + rr) * D_MODEL + h * DK + dv0 + cc] = acc[j];
      }
    }
    __syncthreads();

    // S = S * exp(g_last) + k_dec^T @ v_new ; 8 x NTV tiles
    float eg = __expf(g_last);
    for (int tt2 = wid; tt2 < 8 * NTV; tt2 += 8) {
      int m0 = (tt2 / NTV) * 16, n0 = (tt2 % NTV) * 16;  // m0: Dk, n0: Dv slice
      v8f acc = {0, 0, 0, 0, 0, 0, 0, 0};
      for (int k0 = 0; k0 < CHUNKC; k0 += 32) {
        v16bf kf = make_frag([&](int r, int kk) {
          int c = k0 + kk;
          return kbase[(size_t)(t0 + c) * QKV_COLS + m0 + r] * __expf(g_last - gcs_s[c]);
        });
        v16bf vf = make_frag([&](int nn, int kk) { return vn[(k0 + kk) * DVB + n0 + nn]; });
        acc = wmma_bf16(kf, vf, acc);
      }
#pragma unroll
      for (int j = 0; j < 8; ++j) {
        int rr = m0 + j + 8 * half, cc = n0 + col;
        S[rr * DVB + cc] = S[rr * DVB + cc] * eg + acc[j];
      }
    }
    __syncthreads();
  }
}

// ---------------- gated RMSNorm * silu(z) -> bf16 ----------------
__global__ void __launch_bounds__(128)
gated_norm_kernel(const float* __restrict__ o, const float* __restrict__ qkvz,
                  const float* __restrict__ onw, __hip_bfloat16* __restrict__ out) {
  __shared__ float buf[128];
  int idx = blockIdx.x;
  int t = idx >> 4, h = idx & 15;
  int tid = threadIdx.x;
  const float* orow = o + (size_t)t * D_MODEL + h * DK;
  const float* zrow = qkvz + (size_t)t * QKVZ_COLS + QKV_COLS + h * DK;
  float v = orow[tid];
  buf[tid] = v * v;
  __syncthreads();
  for (int s = 64; s > 0; s >>= 1) {
    if (tid < s) buf[tid] += buf[tid + s];
    __syncthreads();
  }
  float inv = rsqrtf(buf[0] / (float)DK + 1e-5f);
  out[(size_t)t * D_MODEL + h * DK + tid] =
      __float2bfloat16(v * inv * onw[tid] * silu_f(zrow[tid]));
}

// ---------------- host orchestration ----------------
extern "C" void kernel_launch(void* const* d_in, const int* in_sizes, int n_in,
                              void* d_out, int out_size, void* d_ws, size_t ws_size,
                              hipStream_t stream) {
  (void)in_sizes; (void)n_in; (void)out_size; (void)ws_size;
  const float* x        = (const float*)d_in[0];
  const float* mask     = (const float*)d_in[1];
  const float* ln1_w    = (const float*)d_in[2];
  const float* qkvz_w   = (const float*)d_in[3];
  const float* ba_w     = (const float*)d_in[4];
  const float* conv_w   = (const float*)d_in[5];
  const float* A_log    = (const float*)d_in[6];
  const float* dt_bias  = (const float*)d_in[7];
  const float* o_norm_w = (const float*)d_in[8];
  const float* out_w    = (const float*)d_in[9];
  const float* ln2_w    = (const float*)d_in[10];
  const float* gate_w   = (const float*)d_in[11];
  const float* up_w     = (const float*)d_in[12];
  const float* down_w   = (const float*)d_in[13];
  float* out = (float*)d_out;

  char* ws = (char*)d_ws;
  size_t off = 0;
  auto alloc = [&](size_t bytes) -> void* {
    void* p = ws + off;
    off += (bytes + 255) & ~(size_t)255;
    return p;
  };

  const size_t TD = (size_t)T_SEQ * D_MODEL;
  __hip_bfloat16* h_bf  = (__hip_bfloat16*)alloc(TD * 2);
  __hip_bfloat16* wq_bf = (__hip_bfloat16*)alloc((size_t)QKVZ_COLS * D_MODEL * 2);
  __hip_bfloat16* wo_bf = (__hip_bfloat16*)alloc((size_t)D_MODEL * D_MODEL * 2);
  __hip_bfloat16* wg_bf = (__hip_bfloat16*)alloc((size_t)M_FFN * D_MODEL * 2);
  __hip_bfloat16* wu_bf = (__hip_bfloat16*)alloc((size_t)M_FFN * D_MODEL * 2);
  __hip_bfloat16* wd_bf = (__hip_bfloat16*)alloc((size_t)D_MODEL * M_FFN * 2);
  float* qkvz   = (float*)alloc((size_t)T_SEQ * QKVZ_COLS * 4);
  float* qkv    = (float*)alloc((size_t)T_SEQ * QKV_COLS * 4);
  float* g_ws   = (float*)alloc((size_t)T_SEQ * H_HEADS * 4);
  float* beta_ws= (float*)alloc((size_t)T_SEQ * H_HEADS * 4);
  float* gcs_ws = (float*)alloc((size_t)H_HEADS * T_SEQ * 4);
  float* u_ws   = (float*)alloc((size_t)H_HEADS * T_SEQ * DK * 4);
  float* w_ws   = (float*)alloc((size_t)H_HEADS * T_SEQ * DK * 4);
  float* o_ws   = (float*)alloc(TD * 4);
  __hip_bfloat16* og_bf = (__hip_bfloat16*)alloc(TD * 2);
  float* proj   = (float*)alloc(TD * 4);
  float* x2     = (float*)alloc(TD * 4);
  __hip_bfloat16* h2_bf = (__hip_bfloat16*)alloc(TD * 2);
  float* gate_f = (float*)alloc((size_t)T_SEQ * M_FFN * 4);
  float* up_f   = (float*)alloc((size_t)T_SEQ * M_FFN * 4);
  __hip_bfloat16* hu_bf = (__hip_bfloat16*)alloc((size_t)T_SEQ * M_FFN * 2);
  float* down_f = (float*)alloc(TD * 4);

  // 1. convert weights to bf16
  f32_to_bf16_kernel<<<2048, 256, 0, stream>>>(qkvz_w, wq_bf, (size_t)QKVZ_COLS * D_MODEL);
  f32_to_bf16_kernel<<<2048, 256, 0, stream>>>(out_w,  wo_bf, (size_t)D_MODEL * D_MODEL);
  f32_to_bf16_kernel<<<2048, 256, 0, stream>>>(gate_w, wg_bf, (size_t)M_FFN * D_MODEL);
  f32_to_bf16_kernel<<<2048, 256, 0, stream>>>(up_w,   wu_bf, (size_t)M_FFN * D_MODEL);
  f32_to_bf16_kernel<<<2048, 256, 0, stream>>>(down_w, wd_bf, (size_t)D_MODEL * M_FFN);

  // 2. h = rms(x, ln1_w)
  rmsnorm_bf16_kernel<<<T_SEQ, 256, 0, stream>>>(x, ln1_w, h_bf, D_MODEL);

  // 3. qkvz = h @ qkvz_w^T
  gemm_bf16_nt_kernel<<<dim3(QKVZ_COLS / BN, T_SEQ / BM), 256, 0, stream>>>(
      h_bf, wq_bf, qkvz, T_SEQ, QKVZ_COLS, D_MODEL);

  // 4. beta, g
  ba_gate_kernel<<<T_SEQ, 256, 0, stream>>>(h_bf, ba_w, A_log, dt_bias, mask, g_ws, beta_ws);

  // 5. causal conv + silu + l2norm (q pre-scaled)
  conv_norm_kernel<<<T_SEQ, 256, 0, stream>>>(qkvz, conv_w, mask, qkv);

  // 6. per-chunk prep: gcs, u, w
  chunk_prep_kernel<<<dim3(NCHUNK, H_HEADS), 256, 0, stream>>>(
      qkv, g_ws, beta_ws, gcs_ws, u_ws, w_ws);

  // 7. inter-chunk scan (head x Dv-slice)
  size_t scan_smem = (size_t)(128 * DVB + CHUNKC * DVB + CHUNKC * CHUNKC + CHUNKC) * sizeof(float);
  scan_kernel<<<dim3(H_HEADS, DK / DVB), 256, scan_smem, stream>>>(
      qkv, gcs_ws, u_ws, w_ws, o_ws);

  // 8. gated RMSNorm * silu(z)
  gated_norm_kernel<<<T_SEQ * H_HEADS, 128, 0, stream>>>(o_ws, qkvz, o_norm_w, og_bf);

  // 9. out proj + residual
  gemm_bf16_nt_kernel<<<dim3(D_MODEL / BN, T_SEQ / BM), 256, 0, stream>>>(
      og_bf, wo_bf, proj, T_SEQ, D_MODEL, D_MODEL);
  add_f32_kernel<<<2048, 256, 0, stream>>>(x, proj, x2, TD);

  // 10. MLP
  rmsnorm_bf16_kernel<<<T_SEQ, 256, 0, stream>>>(x2, ln2_w, h2_bf, D_MODEL);
  gemm_bf16_nt_kernel<<<dim3(M_FFN / BN, T_SEQ / BM), 256, 0, stream>>>(
      h2_bf, wg_bf, gate_f, T_SEQ, M_FFN, D_MODEL);
  gemm_bf16_nt_kernel<<<dim3(M_FFN / BN, T_SEQ / BM), 256, 0, stream>>>(
      h2_bf, wu_bf, up_f, T_SEQ, M_FFN, D_MODEL);
  silu_mul_bf16_kernel<<<2048, 256, 0, stream>>>(gate_f, up_f, hu_bf, (size_t)T_SEQ * M_FFN);
  gemm_bf16_nt_kernel<<<dim3(D_MODEL / BN, T_SEQ / BM), 256, 0, stream>>>(
      hu_bf, wd_bf, down_f, T_SEQ, D_MODEL, M_FFN);
  add_f32_kernel<<<2048, 256, 0, stream>>>(x2, down_f, out, TD);
}
